// AttentionSuper_86517821214274
// MI455X (gfx1250) — matrix operand order repaired and checked by reading, not compiled, not run on hardware
//
#include <hip/hip_runtime.h>
#include <cstdint>

// ---------------- shapes ----------------
#define Bb   64
#define Hh   10
#define Nn   197
#define DHd  64
#define Cc   640
#define C3   1920
#define Rr   64
#define Tt   30
#define NP   224   // N padded to 7*32 for WMMA K-loops

typedef unsigned short u16;
typedef unsigned int   u32;
typedef __attribute__((ext_vector_type(16))) __bf16         v16bf;
typedef __attribute__((ext_vector_type(8)))  float          v8f;
typedef __attribute__((ext_vector_type(8)))  unsigned short ushort8;

__device__ __forceinline__ u16 f2bf(float f) {
    u32 u = __builtin_bit_cast(u32, f);
    u32 r = u + 0x7FFFu + ((u >> 16) & 1u);   // round-to-nearest-even
    return (u16)(r >> 16);
}
__device__ __forceinline__ float bf2f(u16 h) {
    u32 u = ((u32)h) << 16;
    return __builtin_bit_cast(float, u);
}
__device__ __forceinline__ __bf16 bfraw(u16 h) {
    return __builtin_bit_cast(__bf16, h);
}
__device__ __forceinline__ int perm3_src(int r, int rows) {
    // row r of cat([w[0::3], w[1::3], w[2::3]])
    int n0 = (rows + 2) / 3, n1 = (rows + 1) / 3;
    if (r < n0)      return 3 * r;
    if (r < n0 + n1) return 3 * (r - n0) + 1;
    return 3 * (r - n0 - n1) + 2;
}

// ---------------- small prep kernels ----------------
__global__ void zero_kernel(ushort8* p, long n8) {
    long idx = (long)blockIdx.x * blockDim.x + threadIdx.x;
    if (idx < n8) { ushort8 z = {0,0,0,0,0,0,0,0}; p[idx] = z; }
}

__global__ void cvtx_kernel(const float* __restrict__ x, u16* __restrict__ xb) {
    long idx = (long)blockIdx.x * blockDim.x + threadIdx.x;
    if (idx < (long)Bb * Nn * Cc) xb[idx] = f2bf(x[idx]);
}

// w_eff = perm3(w_qkv + (lora_a @ lora_b)^T)  -> bf16 [1920,640]
__global__ void weff_kernel(const float* __restrict__ wqkv,
                            const float* __restrict__ la,
                            const float* __restrict__ lb,
                            u16* __restrict__ weff) {
    long idx = (long)blockIdx.x * blockDim.x + threadIdx.x;
    if (idx >= (long)C3 * Cc) return;
    int r = (int)(idx / Cc), k = (int)(idx % Cc);
    int src = perm3_src(r, C3);
    float acc = wqkv[(size_t)src * Cc + k];
    for (int j = 0; j < Rr; ++j)
        acc += la[(size_t)k * Rr + j] * lb[(size_t)j * C3 + src];
    weff[idx] = f2bf(acc);
}

__global__ void wproj_kernel(const float* __restrict__ wp, u16* __restrict__ wpj) {
    long idx = (long)blockIdx.x * blockDim.x + threadIdx.x;
    if (idx >= (long)Cc * Cc) return;
    int r = (int)(idx / Cc), k = (int)(idx % Cc);
    wpj[idx] = f2bf(wp[(size_t)perm3_src(r, Cc) * Cc + k]);
}

// relative-position index tables fv/fh, u8 [Nn, NP] (values 0..29)
__global__ void fvfh_kernel(unsigned char* __restrict__ fv, unsigned char* __restrict__ fh) {
    int idx = blockIdx.x * blockDim.x + threadIdx.x;
    if (idx >= Nn * Nn) return;
    int i = idx / Nn, j = idx % Nn;
    int iv = 0, ih = 0;
    if (i > 0 && j > 0) {
        int ri = i - 1, rj = j - 1;
        int dv = rj / 14 - ri / 14;
        int dh = rj % 14 - ri % 14;
        dv = dv > 14 ? 14 : (dv < -14 ? -14 : dv);
        dh = dh > 14 ? 14 : (dh < -14 ? -14 : dh);
        iv = dv + 15; ih = dh + 15;
    }
    fv[(size_t)i * NP + j] = (unsigned char)iv;
    fh[(size_t)i * NP + j] = (unsigned char)ih;
}

// =====================================================================
// GEMM core: block tile 64(M) x 128(N), K-step 32, double-buffered LDS.
// 8 waves: wave = (mr band of 16 rows) x (nh half of 64 cols);
// each wave: 1 A-fragment + 4 batched B-fragments -> 4 back-to-back WMMAs.
// =====================================================================
#define GEMM_BODY(AGLOB, WGLOB, KDIM)                                            \
    __shared__ u16 sA[2][64 * 32];                                               \
    __shared__ u16 sW[2][128 * 32];                                              \
    const int t = threadIdx.x;                                                   \
    const int wid = t >> 5, lane = t & 31, lh = lane >> 4, ll = lane & 15;       \
    const int mr = wid >> 1, nh = wid & 1;                                       \
    const int m0 = blockIdx.x * 64;                                              \
    const int n0 = blockIdx.y * 128;                                             \
    const int sr = t >> 2, ch = t & 3;   /* A staging: 64 rows x 4 chunks */     \
    const int wr = t >> 1, wc = t & 1;   /* W staging: 128 rows x 2 chunks */    \
    const u16* Ag = (AGLOB) + (size_t)(m0 + sr) * (KDIM) + ch * 8;               \
    const u16* Wg = (WGLOB) + (size_t)(n0 + wr) * (KDIM) + wc * 16;              \
    v8f cacc[4];                                                                 \
    _Pragma("unroll")                                                            \
    for (int s = 0; s < 4; ++s) cacc[s] = (v8f){0.f,0.f,0.f,0.f,0.f,0.f,0.f,0.f};\
    ushort8 ra  = *(const ushort8*)(Ag);                                         \
    ushort8 rw0 = *(const ushort8*)(Wg);                                         \
    ushort8 rw1 = *(const ushort8*)(Wg + 8);                                     \
    *(ushort8*)(&sA[0][sr * 32 + ch * 8])      = ra;                             \
    *(ushort8*)(&sW[0][wr * 32 + wc * 16])     = rw0;                            \
    *(ushort8*)(&sW[0][wr * 32 + wc * 16 + 8]) = rw1;                            \
    __syncthreads();                                                             \
    const int KSTEPS = (KDIM) / 32;                                              \
    for (int kk = 0; kk < KSTEPS; ++kk) {                                        \
        const int cur = kk & 1;                                                  \
        if (kk + 1 < KSTEPS) {       /* issue next tile's global loads early */  \
            const int k0 = (kk + 1) * 32;                                        \
            ra  = *(const ushort8*)(Ag + k0);                                    \
            rw0 = *(const ushort8*)(Wg + k0);                                    \
            rw1 = *(const ushort8*)(Wg + k0 + 8);                                \
        }                                                                        \
        if (kk + 2 < KSTEPS) {       /* pull K+2 tile toward L2/L0 */            \
            __builtin_prefetch(Ag + (kk + 2) * 32, 0, 3);                        \
            __builtin_prefetch(Wg + (kk + 2) * 32, 0, 3);                        \
        }                                                                        \
        v16bf a;                                                                 \
        {                                                                        \
            const u16* ap = &sA[cur][(mr * 16 + ll) * 32 + lh * 8];              \
            _Pragma("unroll")                                                    \
            for (int e = 0; e < 8; ++e) {                                        \
                a[e] = bfraw(ap[e]); a[8 + e] = bfraw(ap[16 + e]);               \
            }                                                                    \
        }                                                                        \
        v16bf bfr[4];                                                            \
        _Pragma("unroll")                                                        \
        for (int s = 0; s < 4; ++s) {  /* batch all B-frag LDS loads */          \
            const u16* wp = &sW[cur][(nh * 64 + s * 16 + ll) * 32 + lh * 8];     \
            _Pragma("unroll")                                                    \
            for (int e = 0; e < 16; ++e) bfr[s][e] = bfraw(wp[e + (e >> 3) * 8]); \
        }                                                                        \
        _Pragma("unroll")                                                        \
        for (int s = 0; s < 4; ++s)    /* back-to-back XDL issues */             \
            cacc[s] = __builtin_amdgcn_wmma_f32_16x16x32_bf16(                   \
                          false, a, false, bfr[s], (short)0, cacc[s],            \
                          false, false);                                         \
        if (kk + 1 < KSTEPS) {                                                   \
            const int nxt = cur ^ 1;                                             \
            *(ushort8*)(&sA[nxt][sr * 32 + ch * 8])      = ra;                   \
            *(ushort8*)(&sW[nxt][wr * 32 + wc * 16])     = rw0;                  \
            *(ushort8*)(&sW[nxt][wr * 32 + wc * 16 + 8]) = rw1;                  \
        }                                                                        \
        __syncthreads();                                                         \
    }

// QKV GEMM: C[m,c3] = sum_k x[m,k]*w_eff[c3,k] + bias; scatter into q/k/v [B,H,NP,DH]
__global__ __launch_bounds__(256) void gemm_qkv_kernel(
        const u16* __restrict__ xb, const u16* __restrict__ weff,
        const float* __restrict__ bias,
        u16* __restrict__ qb, u16* __restrict__ kb, u16* __restrict__ vb) {
    GEMM_BODY(xb, weff, Cc)
    #pragma unroll
    for (int s = 0; s < 4; ++s) {
        #pragma unroll
        for (int e = 0; e < 8; ++e) {
            int m   = m0 + mr * 16 + e + 8 * lh;
            int col = n0 + nh * 64 + s * 16 + ll;
            float val = cacc[s][e] + bias[col];
            int b = m / Nn, i = m % Nn;
            int which = col / Cc, h = (col % Cc) / DHd, d = col % DHd;
            u16* dst = (which == 0) ? qb : (which == 1) ? kb : vb;
            dst[(((size_t)b * Hh + h) * NP + i) * DHd + d] = f2bf(val);
        }
    }
}

// Projection GEMM: f32 output [B*N, C]
__global__ __launch_bounds__(256) void gemm_proj_kernel(
        const u16* __restrict__ ctxb, const u16* __restrict__ wpj,
        const float* __restrict__ bias, float* __restrict__ out) {
    GEMM_BODY(ctxb, wpj, Cc)
    #pragma unroll
    for (int s = 0; s < 4; ++s) {
        #pragma unroll
        for (int e = 0; e < 8; ++e) {
            int m   = m0 + mr * 16 + e + 8 * lh;
            int col = n0 + nh * 64 + s * 16 + ll;
            out[(size_t)m * Cc + col] = cacc[s][e] + bias[col];
        }
    }
}

// ---------------- rel-pos bias pre-GEMM: qv/qh [B,H,Nn,30] f32 ----------------
__global__ void qvh_kernel(const u16* __restrict__ qbuf,
                           const float* __restrict__ tkv, const float* __restrict__ tkh,
                           float* __restrict__ qv, float* __restrict__ qh) {
    long idx = (long)blockIdx.x * blockDim.x + threadIdx.x;
    const long total = (long)Bb * Hh * Nn * Tt;
    if (idx >= total) return;
    int  tt  = (int)(idx % Tt);
    long row = idx / Tt;                   // (b*H+h)*Nn + i
    int  i   = (int)(row % Nn);
    long bh  = row / Nn;
    const u16* qr = qbuf + ((size_t)bh * NP + i) * DHd;
    float av = 0.f, ah = 0.f;
    for (int d = 0; d < DHd; ++d) {
        float q = bf2f(qr[d]);
        av += q * tkv[tt * DHd + d];
        ah += q * tkh[tt * DHd + d];
    }
    qv[idx] = av; qh[idx] = ah;
}

// ---------------- fused attention ----------------
__global__ __launch_bounds__(256) void attn_kernel(
        const u16* __restrict__ qbuf, const u16* __restrict__ kbuf, const u16* __restrict__ vbuf,
        const float* __restrict__ qv, const float* __restrict__ qh,
        const unsigned char* __restrict__ fv, const unsigned char* __restrict__ fh,
        const float* __restrict__ tvv, const float* __restrict__ tvh,
        u16* __restrict__ ctx) {
    __shared__ float sS[16 * NP];     // scores -> exp values
    __shared__ u16   sP[16 * NP];     // softmax probs (bf16) for PV WMMA
    __shared__ float sOut[16 * 64];
    __shared__ float sSv[16 * 32];
    __shared__ float sSh[16 * 32];
    __shared__ float sRed[256];
    __shared__ float sMax[16];

    const int t = threadIdx.x;
    const int wid = t >> 5, lane = t & 31, lh = lane >> 4, ll = lane & 15;
    const int qt = blockIdx.x, h = blockIdx.y, b = blockIdx.z;
    const int i0 = qt * 16;
    const size_t bh = (size_t)b * Hh + h;
    const u16* qrow = qbuf + (bh * NP + (i0 + ll)) * DHd;

    // zero rel-V bins (no readers until after phase-2 barriers)
    sSv[t] = 0.f; sSv[t + 256] = 0.f;
    sSh[t] = 0.f; sSh[t + 256] = 0.f;

    // ---- phase 1: S = Q(16x64) @ K^T, 13 j-tiles over 8 waves ----
    // Q fragments invariant across j-tiles: hoist both K-halves.
    v16bf a0, a1;
    {
        const u16* ap = qrow + lh * 8;
        ushort8 u0 = *(const ushort8*)(ap);
        ushort8 u1 = *(const ushort8*)(ap + 16);
        ushort8 u2 = *(const ushort8*)(ap + 32);
        ushort8 u3 = *(const ushort8*)(ap + 48);
        #pragma unroll
        for (int e = 0; e < 8; ++e) {
            a0[e] = bfraw(u0[e]); a0[8 + e] = bfraw(u1[e]);
            a1[e] = bfraw(u2[e]); a1[8 + e] = bfraw(u3[e]);
        }
    }
    for (int jt = wid; jt < 13; jt += 8) {
        const int j0 = jt * 16;
        const u16* kp = kbuf + (bh * NP + (j0 + ll)) * DHd + lh * 8;
        // batch both B-fragment loads, then chain the two WMMAs
        v16bf bf0, bf1;
        {
            ushort8 u0 = *(const ushort8*)(kp);
            ushort8 u1 = *(const ushort8*)(kp + 16);
            ushort8 u2 = *(const ushort8*)(kp + 32);
            ushort8 u3 = *(const ushort8*)(kp + 48);
            #pragma unroll
            for (int e = 0; e < 8; ++e) {
                bf0[e] = bfraw(u0[e]); bf0[8 + e] = bfraw(u1[e]);
                bf1[e] = bfraw(u2[e]); bf1[8 + e] = bfraw(u3[e]);
            }
        }
        v8f c = {0.f,0.f,0.f,0.f,0.f,0.f,0.f,0.f};
        c = __builtin_amdgcn_wmma_f32_16x16x32_bf16(false, a0, false, bf0, (short)0, c, false, false);
        c = __builtin_amdgcn_wmma_f32_16x16x32_bf16(false, a1, false, bf1, (short)0, c, false, false);
        #pragma unroll
        for (int e = 0; e < 8; ++e) sS[(e + 8 * lh) * NP + j0 + ll] = c[e];
    }
    __syncthreads();

    // ---- phase 2: bias + softmax + rel-V bins, 16 threads per row ----
    const int r = t >> 4, cthr = t & 15;
    const int i = i0 + r;
    const unsigned char* fvr = fv + (size_t)i * NP;
    const unsigned char* fhr = fh + (size_t)i * NP;
    const float* qvr = qv + (bh * Nn + i) * Tt;
    const float* qhr = qh + (bh * Nn + i) * Tt;

    float lmax = -3.0e38f;
    if (i < Nn) {
        const float scale = 0.125f;                 // 64^-0.5
        for (int j = cthr; j < Nn; j += 16) {
            float s = (sS[r * NP + j] + qvr[fvr[j]] + qhr[fhr[j]]) * scale;
            sS[r * NP + j] = s;
            lmax = fmaxf(lmax, s);
        }
    }
    sRed[t] = lmax;
    __syncthreads();
    if (cthr == 0) {
        float m = sRed[r * 16];
        #pragma unroll
        for (int u = 1; u < 16; ++u) m = fmaxf(m, sRed[r * 16 + u]);
        sMax[r] = m;
    }
    __syncthreads();
    const float mx = sMax[r];
    float lsum = 0.f;
    if (i < Nn) {
        for (int j = cthr; j < Nn; j += 16) {
            float e = __expf(sS[r * NP + j] - mx);
            sS[r * NP + j] = e;
            lsum += e;
        }
    }
    sRed[t] = lsum;
    __syncthreads();
    if (cthr == 0) {
        float ssum = 0.f;
        #pragma unroll
        for (int u = 0; u < 16; ++u) ssum += sRed[r * 16 + u];
        sMax[r] = 1.f / ssum;      // reuse as inverse sum
    }
    __syncthreads();
    const float inv = sMax[r];
    for (int j = cthr; j < NP; j += 16) {
        float p = (i < Nn && j < Nn) ? sS[r * NP + j] * inv : 0.f;
        sP[r * NP + j] = f2bf(p);
        if (i < Nn && j < Nn) {
            atomicAdd(&sSv[r * 32 + fvr[j]], p);    // ds_add_f32
            atomicAdd(&sSh[r * 32 + fhr[j]], p);
        }
    }
    __syncthreads();

    // ---- phase 3: Out = P(16x224) @ V(224x64), 4 waves x one 16-col d-tile ----
    if (wid < 4) {
        const int dt = wid;
        v8f c = {0.f,0.f,0.f,0.f,0.f,0.f,0.f,0.f};
        for (int ks = 0; ks < 7; ++ks) {
            v16bf a, bf;
            const u16* pp = sP + ll * NP + ks * 32 + lh * 8;
            #pragma unroll
            for (int e = 0; e < 8; ++e) { a[e] = bfraw(pp[e]); a[8 + e] = bfraw(pp[16 + e]); }
            const int dcol = dt * 16 + ll;
            const size_t vbase = (bh * NP + ks * 32 + lh * 8) * DHd + dcol;
            #pragma unroll
            for (int e = 0; e < 16; ++e) bf[e] = bfraw(vbuf[vbase + (size_t)(e + (e >> 3) * 8) * DHd]);
            c = __builtin_amdgcn_wmma_f32_16x16x32_bf16(false, a, false, bf, (short)0, c, false, false);
        }
        #pragma unroll
        for (int e = 0; e < 8; ++e) sOut[(e + 8 * lh) * 64 + dt * 16 + ll] = c[e];
    }
    __syncthreads();

    // ---- phase 4: add bin-sums @ rel-V tables, store ctx bf16 [B,N,C] ----
    #pragma unroll
    for (int u = 0; u < 4; ++u) {
        int idx = t + u * 256;
        int rr = idx >> 6, d = idx & 63;
        float acc = sOut[idx];
        for (int tt = 0; tt < Tt; ++tt)
            acc += sSv[rr * 32 + tt] * tvv[tt * 64 + d]
                 + sSh[rr * 32 + tt] * tvh[tt * 64 + d];
        int ii = i0 + rr;
        if (ii < Nn) ctx[((size_t)b * Nn + ii) * Cc + h * DHd + d] = f2bf(acc);
    }
}

// ---------------- host launch ----------------
extern "C" void kernel_launch(void* const* d_in, const int* in_sizes, int n_in,
                              void* d_out, int out_size, void* d_ws, size_t ws_size,
                              hipStream_t stream) {
    (void)in_sizes; (void)n_in; (void)out_size; (void)ws_size;
    const float* x      = (const float*)d_in[0];
    const float* w_qkv  = (const float*)d_in[1];
    const float* b_qkv  = (const float*)d_in[2];
    const float* lora_a = (const float*)d_in[3];
    const float* lora_b = (const float*)d_in[4];
    const float* tkv    = (const float*)d_in[5];
    const float* tkh    = (const float*)d_in[6];
    const float* tvv    = (const float*)d_in[7];
    const float* tvh    = (const float*)d_in[8];
    const float* w_proj = (const float*)d_in[9];
    const float* b_proj = (const float*)d_in[10];
    float* out = (float*)d_out;

    char* ws = (char*)d_ws;
    size_t off = 0;
    auto take = [&](size_t bytes) -> char* {
        char* p = ws + off;
        off = (off + bytes + 255) & ~(size_t)255;
        return p;
    };

    const size_t QKVB = (size_t)Bb * Hh * NP * DHd * 2;       // bf16 q/k/v (padded)
    u16* xb   = (u16*)take((size_t)Bb * Nn * Cc * 2);
    u16* weff = (u16*)take((size_t)C3 * Cc * 2);
    u16* wpj  = (u16*)take((size_t)Cc * Cc * 2);
    u16* qb   = (u16*)take(QKVB);
    u16* kb   = (u16*)take(QKVB);
    u16* vb   = (u16*)take(QKVB);
    float* qv = (float*)take((size_t)Bb * Hh * Nn * Tt * 4);
    float* qh = (float*)take((size_t)Bb * Hh * Nn * Tt * 4);
    unsigned char* fv = (unsigned char*)take((size_t)Nn * NP);
    unsigned char* fh = (unsigned char*)take((size_t)Nn * NP);
    u16* ctxb = (u16*)take((size_t)Bb * Nn * Cc * 2);

    const long z8 = (long)(QKVB / 16);
    zero_kernel<<<(int)((z8 + 255) / 256), 256, 0, stream>>>((ushort8*)qb, z8);
    zero_kernel<<<(int)((z8 + 255) / 256), 256, 0, stream>>>((ushort8*)kb, z8);
    zero_kernel<<<(int)((z8 + 255) / 256), 256, 0, stream>>>((ushort8*)vb, z8);

    {
        long n = (long)Bb * Nn * Cc;
        cvtx_kernel<<<(int)((n + 255) / 256), 256, 0, stream>>>(x, xb);
    }
    weff_kernel<<<(int)(((long)C3 * Cc + 255) / 256), 256, 0, stream>>>(w_qkv, lora_a, lora_b, weff);
    wproj_kernel<<<(int)(((long)Cc * Cc + 255) / 256), 256, 0, stream>>>(w_proj, wpj);
    fvfh_kernel<<<(Nn * Nn + 255) / 256, 256, 0, stream>>>(fv, fh);

    gemm_qkv_kernel<<<dim3((Bb * Nn) / 64, C3 / 128), 256, 0, stream>>>(xb, weff, b_qkv, qb, kb, vb);

    {
        long n = (long)Bb * Hh * Nn * Tt;
        qvh_kernel<<<(int)((n + 255) / 256), 256, 0, stream>>>(qb, tkv, tkh, qv, qh);
    }

    attn_kernel<<<dim3(13, Hh, Bb), 256, 0, stream>>>(qb, kb, vb, qv, qh, fv, fh, tvv, tvh, ctxb);

    gemm_proj_kernel<<<dim3((Bb * Nn) / 64, Cc / 128), 256, 0, stream>>>(ctxb, wpj, b_proj, out);
}
